// SynthesisLayer_9216999817688
// MI455X (gfx1250) — compile-verified, matrix operand order checked by reading, not run on hardware
//
#include <hip/hip_runtime.h>
#include <hip/hip_bf16.h>
#include <stdint.h>

#define B_    16
#define IC_   512
#define OC_   512
#define HIN   32
#define RES_  64
#define KTOT  4608      // 9 taps * 512 ic
#define NKSTEP 144      // KTOT / 32
#define XPAD  34
#define LDP   40        // LDS row pitch in bf16 units (80B) -> conflict-free
#define TILE_HALF 5120  // 128 * LDP

typedef __attribute__((ext_vector_type(16))) __bf16 v16bf;
typedef __attribute__((ext_vector_type(8)))  float  v8f;

union FragU { uint4 q[2]; v16bf v; };

// ---------- prep: sum of squared conv weights per (oc, ic) ----------
__global__ void wsq_kernel(const float* __restrict__ cw, float* __restrict__ wsq) {
  int i = blockIdx.x * 256 + threadIdx.x;          // oc*512 + ic
  const float* p = cw + (size_t)i * 9;
  float s = 0.f;
#pragma unroll
  for (int k = 0; k < 9; ++k) { float v = p[k]; s += v * v; }
  wsq[i] = s;
}

// ---------- styles = w @ (aw * 1/sqrt(512))^T + bias ----------
__global__ void styles_kernel(const float* __restrict__ w, const float* __restrict__ aw,
                              const float* __restrict__ ab, float* __restrict__ st) {
  int i = blockIdx.x * 256 + threadIdx.x;          // b*512 + ic
  int ic = i & 511, b = i >> 9;
  const float* wr = w + (size_t)b * 512;
  const float* ar = aw + (size_t)ic * 512;
  float acc = 0.f;
  for (int k = 0; k < 512; ++k) acc += wr[k] * ar[k];
  st[i] = acc * 0.04419417382415922f + ab[ic];
}

// ---------- dcoefs[b][oc] = rsqrt( sum_ic styles^2 * wsq + 1e-8 ) ----------
__global__ void dcoef_kernel(const float* __restrict__ st, const float* __restrict__ wsq,
                             float* __restrict__ dc) {
  int i = blockIdx.x * 256 + threadIdx.x;          // b*512 + oc
  int oc = i & 511, b = i >> 9;
  const float* sr = st + (size_t)b * 512;
  const float* qr = wsq + (size_t)oc * 512;
  float acc = 0.f;
  for (int k = 0; k < 512; ++k) { float s = sr[k]; acc += s * s * qr[k]; }
  dc[i] = rsqrtf(acc + 1e-8f);
}

__global__ void zero_kernel(uint32_t* p, int n) {
  int i = blockIdx.x * 256 + threadIdx.x;
  if (i < n) p[i] = 0u;
}

// ---------- x (fp32 NCHW) -> bf16 padded K-major [b][y][x][ic] ----------
__global__ void xT_kernel(const float* __restrict__ x, __hip_bfloat16* __restrict__ xT) {
  __shared__ float tile[32][33];
  int b = blockIdx.z, ys = blockIdx.y, ic0 = blockIdx.x * 32;
  int tx = threadIdx.x & 31, ty = threadIdx.x >> 5;
#pragma unroll
  for (int i = 0; i < 4; ++i) {
    int ic = ic0 + ty + i * 8;
    tile[ty + i * 8][tx] = x[(((size_t)b * 512 + ic) * 32 + ys) * 32 + tx];
  }
  __syncthreads();
#pragma unroll
  for (int i = 0; i < 4; ++i) {
    int xs = ty + i * 8;
    xT[(((size_t)b * XPAD + (ys + 1)) * XPAD + (xs + 1)) * 512 + ic0 + tx] =
        __float2bfloat16(tile[tx][xs]);
  }
}

// ---------- effective weights: Geff[b][r][oc][tap][ic]  (bf16) ----------
// Composite of modulation * demod and the FIR folded through the stride-2
// transposed conv: per output parity (ru,rv), a 3x3 kernel over x.
__global__ void geff_kernel(const float* __restrict__ cw, const float* __restrict__ st,
                            const float* __restrict__ dc, const float* __restrict__ f,
                            __hip_bfloat16* __restrict__ G) {
  int i = blockIdx.x * 256 + threadIdx.x;          // (b, oc, ic) flat, ic fastest
  int ic = i & 511, oc = (i >> 9) & 511, b = i >> 18;
  float w9[9];
  const float* p = cw + ((size_t)oc * 512 + ic) * 9;
#pragma unroll
  for (int k = 0; k < 9; ++k) w9[k] = p[k];
  float sd = 4.0f * st[b * 512 + ic] * dc[b * 512 + oc];  // 4 = UP^2 FIR gain
  const int mt0[3] = {2, 0, -2};   // ru/rv = 0 (even outputs)
  const int mt1[3] = {3, 1, -1};   // ru/rv = 1 (odd outputs)
  for (int r = 0; r < 4; ++r) {
    int ru = r >> 1, rv = r & 1;
    size_t base = (((size_t)(b * 4 + r) * 512 + oc) * (size_t)KTOT) + ic;
    for (int di = 0; di < 3; ++di)
      for (int dj = 0; dj < 3; ++dj) {
        int mi = ru ? mt1[di] : mt0[di];
        int mj = rv ? mt1[dj] : mt0[dj];
        float g = 0.f;
#pragma unroll
        for (int kh = 0; kh < 3; ++kh) {
          int s_ = kh - mi + 1;
          if (s_ < 0 || s_ > 3) continue;
#pragma unroll
          for (int kw = 0; kw < 3; ++kw) {
            int t_ = kw - mj + 1;
            if (t_ < 0 || t_ > 3) continue;
            g += w9[kh * 3 + kw] * f[s_ * 4 + t_];
          }
        }
        G[base + (size_t)(di * 3 + dj) * 512] = __float2bfloat16(g * sd);
      }
  }
}

// issue one 16B async global->LDS copy (GV form, tracked by ASYNCcnt)
__device__ __forceinline__ void async_b128(uint32_t lds_off, uint64_t gaddr) {
  asm volatile("global_load_async_to_lds_b128 %0, %1, off"
               :: "v"(lds_off), "v"(gaddr) : "memory");
}
__device__ __forceinline__ void wait_async0() {
  asm volatile("s_wait_asynccnt 0" ::: "memory");
}

// ---------- main implicit GEMM: per (b, parity) C[512 oc x 1024 n], K=4608 ----------
__global__ __launch_bounds__(256)
void modconv_wmma_kernel(const __hip_bfloat16* __restrict__ G,
                         const __hip_bfloat16* __restrict__ xT,
                         const float* __restrict__ noise,
                         const float* __restrict__ bias,
                         float* __restrict__ out) {
  __shared__ __align__(16) unsigned short smem[4 * TILE_HALF];  // 40 KB, dbl-buffered
  const int tid = threadIdx.x;
  const int blk_m = blockIdx.x & 3, blk_n = blockIdx.x >> 2;
  const int r = blockIdx.y, b = blockIdx.z;
  const int ru = r >> 1, rv = r & 1;
  const int m0 = blk_m * 128;
  const int p0 = blk_n * 4;  // n-tile = 4 p-rows x 32 q

  const __hip_bfloat16* Ab = G + ((size_t)(b * 4 + r) * OC_) * KTOT;
  const __hip_bfloat16* Xb = xT + (size_t)b * XPAD * XPAD * 512;

  // staging: 512 16B-chunks per tile, 2 per thread per tile
  const int rowA0 = tid >> 2;          // 0..63
  const int rowA1 = rowA0 + 64;        // 64..127
  const int coff  = (tid & 3) * 8;     // bf16 offset inside 32-k chunk
  const int pB0 = p0 + (rowA0 >> 5), qB0 = rowA0 & 31;
  const int pB1 = p0 + (rowA1 >> 5), qB1 = rowA1 & 31;
  const __hip_bfloat16* a0p = Ab + (size_t)(m0 + rowA0) * KTOT + coff;
  const __hip_bfloat16* a1p = Ab + (size_t)(m0 + rowA1) * KTOT + coff;

  // LDS byte offsets for this thread's 4 destination chunks (per buffer)
  const uint32_t smemBase = (uint32_t)(uintptr_t)&smem[0];   // addr[31:0] = LDS offset
  const uint32_t dA0 = smemBase + (uint32_t)(rowA0 * LDP + coff) * 2;
  const uint32_t dA1 = smemBase + (uint32_t)(rowA1 * LDP + coff) * 2;
  const uint32_t dB0 = dA0 + TILE_HALF * 2;
  const uint32_t dB1 = dA1 + TILE_HALF * 2;
  const uint32_t bufStride = 2 * TILE_HALF * 2;              // bytes per buffer

  auto issueAsync = [&](int kt, int s) {
    const int k0 = kt << 5;
    const int tap = kt >> 4;            // 16 k-steps per tap
    const int di = tap / 3, dj = tap - di * 3;
    const int icc = (kt & 15) << 5;
    const uint32_t bo = (uint32_t)s * bufStride;
    async_b128(dA0 + bo, (uint64_t)(uintptr_t)(a0p + k0));
    async_b128(dA1 + bo, (uint64_t)(uintptr_t)(a1p + k0));
    async_b128(dB0 + bo, (uint64_t)(uintptr_t)(
        Xb + ((size_t)(pB0 + di) * XPAD + (qB0 + dj)) * 512 + icc + coff));
    async_b128(dB1 + bo, (uint64_t)(uintptr_t)(
        Xb + ((size_t)(pB1 + di) * XPAD + (qB1 + dj)) * 512 + icc + coff));
  };

  const int lane = tid & 31;
  const int wv = tid >> 5;
  const int wm = wv & 3, wn = wv >> 2;          // wave grid 4(M) x 2(N)
  const int laneN = lane & 15, laneHi = lane >> 4;

  v8f acc[2][4];
  {
    v8f z = {};
    for (int i = 0; i < 2; ++i)
      for (int j = 0; j < 4; ++j) acc[i][j] = z;
  }

  auto compute = [&](int s) {
    const unsigned short* As = smem + s * 2 * TILE_HALF;
    const unsigned short* Bs = As + TILE_HALF;
    FragU fa[2], fb[4];
#pragma unroll
    for (int fm = 0; fm < 2; ++fm) {
      const unsigned short* p = &As[(wm * 32 + fm * 16 + laneN) * LDP + laneHi * 8];
      fa[fm].q[0] = *reinterpret_cast<const uint4*>(p);       // K 0-7 / 8-15
      fa[fm].q[1] = *reinterpret_cast<const uint4*>(p + 16);  // K 16-23 / 24-31
    }
#pragma unroll
    for (int fn = 0; fn < 4; ++fn) {
      const unsigned short* p = &Bs[(wn * 64 + fn * 16 + laneN) * LDP + laneHi * 8];
      fb[fn].q[0] = *reinterpret_cast<const uint4*>(p);
      fb[fn].q[1] = *reinterpret_cast<const uint4*>(p + 16);
    }
#pragma unroll
    for (int fm = 0; fm < 2; ++fm)
#pragma unroll
      for (int fn = 0; fn < 4; ++fn)
        acc[fm][fn] = __builtin_amdgcn_wmma_f32_16x16x32_bf16(
            false, fa[fm].v, false, fb[fn].v, (short)0, acc[fm][fn], false, false);
  };

  // software pipeline: wait(buf s) -> barrier -> issue(buf s^1) -> compute(buf s)
  issueAsync(0, 0);
  for (int kt = 0; kt < NKSTEP; ++kt) {
    const int s = kt & 1;
    wait_async0();        // our batch for buf s has landed in LDS
    __syncthreads();      // all waves' batches landed; all done reading buf s^1
    if (kt + 1 < NKSTEP) issueAsync(kt + 1, s ^ 1);
    compute(s);
  }

  // fused epilogue: noise + bias + leaky_relu * sqrt(2), strided parity store
  const float gain = 1.41421356237309515f;
#pragma unroll
  for (int fm = 0; fm < 2; ++fm) {
#pragma unroll
    for (int fn = 0; fn < 4; ++fn) {
      int nloc = blk_n * 128 + wn * 64 + fn * 16 + laneN;
      int pp = nloc >> 5, qq = nloc & 31;
      int oy = 2 * pp + ru, ox = 2 * qq + rv;
      float nz = noise[oy * RES_ + ox];
#pragma unroll
      for (int rr = 0; rr < 8; ++rr) {
        int oc = m0 + wm * 32 + fm * 16 + rr + laneHi * 8;
        float v = acc[fm][fn][rr] + nz + bias[oc];
        v = (v > 0.f ? v : 0.2f * v) * gain;
        out[(((size_t)b * OC_ + oc) * RES_ + oy) * RES_ + ox] = v;
      }
    }
  }
}

extern "C" void kernel_launch(void* const* d_in, const int* in_sizes, int n_in,
                              void* d_out, int out_size, void* d_ws, size_t ws_size,
                              hipStream_t stream) {
  const float* x     = (const float*)d_in[0];
  const float* w     = (const float*)d_in[1];
  const float* aw    = (const float*)d_in[2];
  const float* ab    = (const float*)d_in[3];
  const float* cw    = (const float*)d_in[4];
  const float* bias  = (const float*)d_in[5];
  const float* noise = (const float*)d_in[6];
  const float* f     = (const float*)d_in[7];
  float* out = (float*)d_out;

  char* ws = (char*)d_ws;
  float* styles = (float*)(ws);                        // 32 KB
  float* dcoef  = (float*)(ws + 32768);                // 32 KB
  float* wsq    = (float*)(ws + 65536);                // 1 MB
  __hip_bfloat16* xT = (__hip_bfloat16*)(ws + 1114112);   // 18.9 MB
  __hip_bfloat16* G  = (__hip_bfloat16*)(ws + 20054016);  // 302 MB

  wsq_kernel<<<1024, 256, 0, stream>>>(cw, wsq);
  styles_kernel<<<32, 256, 0, stream>>>(w, aw, ab, styles);
  dcoef_kernel<<<32, 256, 0, stream>>>(styles, wsq, dcoef);
  const int zn = (16 * XPAD * XPAD * 512 * 2) / 4;
  zero_kernel<<<(zn + 255) / 256, 256, 0, stream>>>((uint32_t*)xT, zn);
  xT_kernel<<<dim3(16, 32, 16), 256, 0, stream>>>(x, xT);
  geff_kernel<<<16384, 256, 0, stream>>>(cw, styles, dcoef, f, G);
  modconv_wmma_kernel<<<dim3(32, 4, 16), 256, 0, stream>>>(G, xT, noise, bias, out);
}
